// SpatialLayer_29669634081261
// MI455X (gfx1250) — matrix operand order repaired and checked by reading, not compiled, not run on hardware
//
#include <hip/hip_runtime.h>
#include <hip/hip_bf16.h>

typedef __bf16 bf16;
typedef __attribute__((ext_vector_type(16))) __bf16 v16bf;
typedef __attribute__((ext_vector_type(8)))  __bf16 v8bf;
typedef __attribute__((ext_vector_type(8)))  float  v8f;

#define HID 128
#define HEADS 4
#define CH 32
#define RPB 5          // row tiles per block in the GEMM (3125 = 5*625)

static __device__ __forceinline__ void atomicMaxF(float* addr, float val) {
    unsigned int* ua = (unsigned int*)addr;
    unsigned int old = *ua;
    while (__uint_as_float(old) < val) {
        unsigned int assumed = old;
        old = atomicCAS(ua, assumed, __float_as_uint(val));
        if (old == assumed) break;
    }
}

// ---------------- conversion ----------------
__global__ void k_f32_to_bf16(const float* __restrict__ in, bf16* __restrict__ out, int n) {
    int i = blockIdx.x * blockDim.x + threadIdx.x;
    if (i < n) out[i] = (bf16)in[i];
}

// ---------------- fills ----------------
__global__ void k_fill(float* __restrict__ p, float v, int n) {
    int i = blockIdx.x * blockDim.x + threadIdx.x;
    if (i < n) p[i] = v;
}

// ---------------- weight pack: f32 [128x128] row-major -> bf16 WMMA B-lane layout ----------------
__global__ void k_pack_B(const float* __restrict__ W, bf16* __restrict__ P) {
    int idx = blockIdx.x * blockDim.x + threadIdx.x;
    if (idx >= HID * HID) return;
    int i    = idx & 15;
    int lane = (idx >> 4) & 31;
    int kt   = (idx >> 9) & 3;
    int ct   = (idx >> 11) & 7;
    int k    = kt * 32 + (lane >> 4) * 16 + i;
    int col  = ct * 16 + (lane & 15);
    P[idx] = (bf16)W[(size_t)k * HID + col];
}

// ---------------- WMMA bf16 GEMM: C[N x 128] = A[N x 128] @ B[128 x 128] (+bias) ----------------
// block = 256 (8 waves); wave w owns column tile w. B held in registers (4 k-tiles),
// reused across RPB row tiles. A/C addressed via per-lane pointers bumped by 16*HID per tile,
// all element offsets are compile-time immediates.
__global__ __launch_bounds__(256) void k_gemm_bf16_128(
    const bf16* __restrict__ A, const bf16* __restrict__ Bp,
    const float* __restrict__ bias, float* __restrict__ C, int ntiles)
{
    const int lane = threadIdx.x & 31;
    const int wave = threadIdx.x >> 5;
    const int mrow = lane & 15;
    const int hi   = lane >> 4;

    // B: packed layout, 16 contiguous bf16 per (col-tile, k-tile, lane)
    const bf16* bp = Bp + (size_t)wave * 2048 + (size_t)lane * 16;
    v16bf bm0 = *(const v16bf*)(bp);
    v16bf bm1 = *(const v16bf*)(bp + 512);
    v16bf bm2 = *(const v16bf*)(bp + 1024);
    v16bf bm3 = *(const v16bf*)(bp + 1536);

    const int col = wave * 16 + mrow;
    const float bb = bias ? bias[col] : 0.0f;

    int t0   = blockIdx.x * RPB;
    int tend = t0 + RPB; if (tend > ntiles) tend = ntiles;

    // Per-lane streaming pointers (strength-reduced addressing).
    const bf16* arow = A + (size_t)(t0 * 16 + mrow) * HID + hi * 16;
    float*      cp   = C + (size_t)(t0 * 16 + hi * 8) * HID + col;

    for (int tile = t0; tile < tend; ++tile) {
        union { v16bf v; v8bf h[2]; } am;
        v8f acc = {};

        am.h[0] = *(const v8bf*)(arow +  0); am.h[1] = *(const v8bf*)(arow + 8);
        acc = __builtin_amdgcn_wmma_f32_16x16x32_bf16(false, am.v, false, bm0, (short)0, acc, false, false);
        am.h[0] = *(const v8bf*)(arow + 32); am.h[1] = *(const v8bf*)(arow + 40);
        acc = __builtin_amdgcn_wmma_f32_16x16x32_bf16(false, am.v, false, bm1, (short)0, acc, false, false);
        am.h[0] = *(const v8bf*)(arow + 64); am.h[1] = *(const v8bf*)(arow + 72);
        acc = __builtin_amdgcn_wmma_f32_16x16x32_bf16(false, am.v, false, bm2, (short)0, acc, false, false);
        am.h[0] = *(const v8bf*)(arow + 96); am.h[1] = *(const v8bf*)(arow + 104);
        acc = __builtin_amdgcn_wmma_f32_16x16x32_bf16(false, am.v, false, bm3, (short)0, acc, false, false);

        // C/D layout: VGPR r -> row (r + 8*hi); stores use immediate offsets r*512B
#pragma unroll
        for (int r = 0; r < 8; ++r)
            cp[r * HID] = acc[r] + bb;

        arow += 16 * HID;
        cp   += 16 * HID;
    }
}

// ---------------- GCN normalization ----------------
__global__ void k_deg_accum(const int* __restrict__ tgt, float* __restrict__ deg, int E) {
    int e = blockIdx.x * blockDim.x + threadIdx.x;
    if (e < E) atomicAdd(deg + tgt[e], 1.0f);
}
__global__ void k_rsqrt_inplace(float* __restrict__ d, int n) {
    int i = blockIdx.x * blockDim.x + threadIdx.x;
    if (i < n) { float v = d[i]; d[i] = (v > 0.0f) ? rsqrtf(v) : 0.0f; }
}
// h[t] += h0[s] * dinv[s]*dinv[t]; one thread per (edge, 4 features)
__global__ void k_aggregate(const int* __restrict__ src, const int* __restrict__ tgt,
                            const float* __restrict__ dinv, const float* __restrict__ h0,
                            float* __restrict__ h, int E)
{
    int idx = blockIdx.x * blockDim.x + threadIdx.x;
    if (idx >= E * 32) return;
    int e = idx >> 5;
    int c = (idx & 31) * 4;
    int s = src[e], t = tgt[e];
    float norm = dinv[s] * dinv[t];
    float4 hv = *(const float4*)(h0 + (size_t)s * HID + c);
    float* op = h + (size_t)t * HID + c;
    atomicAdd(op + 0, hv.x * norm);
    atomicAdd(op + 1, hv.y * norm);
    atomicAdd(op + 2, hv.z * norm);
    atomicAdd(op + 3, hv.w * norm);
}
// h = relu(h_agg + self_loop + bias)
__global__ void k_finalize_h(const float* __restrict__ h0, const float* __restrict__ dinv,
                             const float* __restrict__ bg, float* __restrict__ h, int n128)
{
    int idx = blockIdx.x * blockDim.x + threadIdx.x;
    if (idx >= n128) return;
    int i = idx >> 7, f = idx & 127;
    float di = dinv[i];
    float v = h[idx] + h0[idx] * di * di + bg[f];
    h[idx] = fmaxf(v, 0.0f);
}

// ---------------- TransformerConv attention ----------------
__global__ void k_alpha(const int* __restrict__ src, const int* __restrict__ tgt,
                        const float* __restrict__ q, const float* __restrict__ kk,
                        float* __restrict__ alpha, float* __restrict__ amax, int E)
{
    int idx = blockIdx.x * blockDim.x + threadIdx.x;
    if (idx >= E * HEADS) return;
    int e = idx >> 2, hh = idx & 3;
    int s = src[e], t = tgt[e];
    const float4* qp = (const float4*)(q  + (size_t)t * HID + hh * CH);
    const float4* kp = (const float4*)(kk + (size_t)s * HID + hh * CH);
    float a = 0.0f;
#pragma unroll
    for (int j = 0; j < 8; ++j) {
        float4 qv = qp[j], kv = kp[j];
        a += qv.x * kv.x + qv.y * kv.y + qv.z * kv.z + qv.w * kv.w;
    }
    a *= 0.17677669529663687f;   // 1/sqrt(32)
    alpha[idx] = a;
    atomicMaxF(amax + (size_t)t * HEADS + hh, a);
}
__global__ void k_expsum(const int* __restrict__ tgt, const float* __restrict__ amax,
                         float* __restrict__ alpha, float* __restrict__ denom, int E)
{
    int idx = blockIdx.x * blockDim.x + threadIdx.x;
    if (idx >= E * HEADS) return;
    int e = idx >> 2, hh = idx & 3;
    int t = tgt[e];
    float ea = __expf(alpha[idx] - amax[(size_t)t * HEADS + hh]);
    alpha[idx] = ea;
    atomicAdd(denom + (size_t)t * HEADS + hh, ea);
}
__global__ void k_attn_out(const int* __restrict__ src, const int* __restrict__ tgt,
                           const float* __restrict__ ea, const float* __restrict__ denom,
                           const float* __restrict__ vv, float* __restrict__ out, int E)
{
    int idx = blockIdx.x * blockDim.x + threadIdx.x;
    if (idx >= E * 32) return;
    int e = idx >> 5;
    int c4 = idx & 31;
    int c = c4 * 4;
    int hh = c4 >> 3;
    int s = src[e], t = tgt[e];
    float d = denom[(size_t)t * HEADS + hh];
    float w = ea[(size_t)e * HEADS + hh] / ((d > 0.0f) ? d : 1.0f);
    float4 vvl = *(const float4*)(vv + (size_t)s * HID + c);
    float* op = out + (size_t)t * HID + c;
    atomicAdd(op + 0, vvl.x * w);
    atomicAdd(op + 1, vvl.y * w);
    atomicAdd(op + 2, vvl.z * w);
    atomicAdd(op + 3, vvl.w * w);
}

static inline int ceil_div(long long a, int b) { return (int)((a + b - 1) / b); }

extern "C" void kernel_launch(void* const* d_in, const int* in_sizes, int n_in,
                              void* d_out, int out_size, void* d_ws, size_t ws_size,
                              hipStream_t stream)
{
    const float* x     = (const float*)d_in[0];
    const int*   eidx  = (const int*)d_in[1];
    const float* Wgcn  = (const float*)d_in[2];
    const float* bgcn  = (const float*)d_in[3];
    const float* Wq    = (const float*)d_in[4];
    const float* bq    = (const float*)d_in[5];
    const float* Wk    = (const float*)d_in[6];
    const float* bk    = (const float*)d_in[7];
    const float* Wv    = (const float*)d_in[8];
    const float* bv    = (const float*)d_in[9];
    const float* Ws    = (const float*)d_in[10];
    const float* bs    = (const float*)d_in[11];

    const int N = in_sizes[0] / HID;
    const int E = in_sizes[1] / 2;
    const int* src = eidx;
    const int* tgt = eidx + E;

    // ---- workspace carve-up (256B aligned) ----
    char* base = (char*)d_ws;
    size_t off = 0;
    auto carve = [&](size_t bytes) -> char* {
        char* p = base + off;
        off = (off + bytes + 255) & ~(size_t)255;
        return p;
    };
    bf16*  xb    = (bf16*)carve((size_t)N * HID * sizeof(bf16));   // also reused as hb
    bf16*  wb    = (bf16*)carve(5 * (size_t)HID * HID * sizeof(bf16)); // packed weights
    float* h0    = (float*)carve((size_t)N * HID * sizeof(float)); // reused as q
    float* h     = (float*)carve((size_t)N * HID * sizeof(float));
    float* kmat  = (float*)carve((size_t)N * HID * sizeof(float));
    float* vmat  = (float*)carve((size_t)N * HID * sizeof(float));
    float* deg   = (float*)carve((size_t)N * sizeof(float));       // becomes dinv
    float* amax  = (float*)carve((size_t)N * HEADS * sizeof(float));
    float* denom = (float*)carve((size_t)N * HEADS * sizeof(float));
    float* alpha = (float*)carve((size_t)E * HEADS * sizeof(float)); // becomes ea
    (void)ws_size; (void)n_in; (void)out_size;

    bf16* wb_gcn = wb + 0 * HID * HID;
    bf16* wb_q   = wb + 1 * HID * HID;
    bf16* wb_k   = wb + 2 * HID * HID;
    bf16* wb_v   = wb + 3 * HID * HID;
    bf16* wb_s   = wb + 4 * HID * HID;

    const int nf = N * HID;
    const int T = 256;
    const int ntiles = N / 16;                 // 3125
    const int gblocks = ceil_div(ntiles, RPB); // 625

    // 1) convert x; pack all weight matrices into WMMA B layout (fused f32->bf16)
    k_f32_to_bf16<<<ceil_div(nf, T), T, 0, stream>>>(x, xb, nf);
    k_pack_B<<<ceil_div(HID * HID, T), T, 0, stream>>>(Wgcn, wb_gcn);
    k_pack_B<<<ceil_div(HID * HID, T), T, 0, stream>>>(Wq, wb_q);
    k_pack_B<<<ceil_div(HID * HID, T), T, 0, stream>>>(Wk, wb_k);
    k_pack_B<<<ceil_div(HID * HID, T), T, 0, stream>>>(Wv, wb_v);
    k_pack_B<<<ceil_div(HID * HID, T), T, 0, stream>>>(Ws, wb_s);

    // 2) h0 = x @ Wgcn (WMMA)
    k_gemm_bf16_128<<<dim3(gblocks), T, 0, stream>>>(xb, wb_gcn, nullptr, h0, ntiles);

    // 3) symmetric normalization with self-loops: deg init = 1 (self loop)
    k_fill<<<ceil_div(N, T), T, 0, stream>>>(deg, 1.0f, N);
    k_deg_accum<<<ceil_div(E, T), T, 0, stream>>>(tgt, deg, E);
    k_rsqrt_inplace<<<ceil_div(N, T), T, 0, stream>>>(deg, N);   // deg -> dinv

    // 4) aggregate: h = sum_e norm * h0[src] ; then + self loop + bias, relu
    k_fill<<<ceil_div(nf, T), T, 0, stream>>>(h, 0.0f, nf);
    k_aggregate<<<ceil_div((long long)E * 32, T), T, 0, stream>>>(src, tgt, deg, h0, h, E);
    k_finalize_h<<<ceil_div(nf, T), T, 0, stream>>>(h0, deg, bgcn, h, nf);

    // 5) q,k,v,skip projections (WMMA); hb reuses xb, q reuses h0, skip -> d_out
    bf16* hb = xb;
    float* q = h0;
    float* outp = (float*)d_out;
    k_f32_to_bf16<<<ceil_div(nf, T), T, 0, stream>>>(h, hb, nf);
    k_gemm_bf16_128<<<dim3(gblocks), T, 0, stream>>>(hb, wb_q, bq, q, ntiles);
    k_gemm_bf16_128<<<dim3(gblocks), T, 0, stream>>>(hb, wb_k, bk, kmat, ntiles);
    k_gemm_bf16_128<<<dim3(gblocks), T, 0, stream>>>(hb, wb_v, bv, vmat, ntiles);
    k_gemm_bf16_128<<<dim3(gblocks), T, 0, stream>>>(hb, wb_s, bs, outp, ntiles); // skip

    // 6) attention softmax over incoming edges
    k_fill<<<ceil_div(N * HEADS, T), T, 0, stream>>>(amax, -3.0e38f, N * HEADS);
    k_fill<<<ceil_div(N * HEADS, T), T, 0, stream>>>(denom, 0.0f, N * HEADS);
    k_alpha<<<ceil_div((long long)E * HEADS, T), T, 0, stream>>>(src, tgt, q, kmat, alpha, amax, E);
    k_expsum<<<ceil_div((long long)E * HEADS, T), T, 0, stream>>>(tgt, amax, alpha, denom, E);

    // 7) out += softmax-weighted V aggregation
    k_attn_out<<<ceil_div((long long)E * 32, T), T, 0, stream>>>(src, tgt, alpha, denom, vmat, outp, E);
}
